// AttLoc_27401891349075
// MI455X (gfx1250) — matrix operand order repaired and checked by reading, not compiled, not run on hardware
//
#include <hip/hip_runtime.h>
#include <hip/hip_bf16.h>
#include <math.h>

#define B_      32
#define T_      2000
#define EPROJS_ 1024
#define DUNITS_ 1024
#define ATT_    512
#define CCH_    10
#define KF_     201
#define PADF_   100
#define LDSROW  1032   // 1024 + 8 ushort pad -> conflict-free ds_read_b128

typedef __attribute__((ext_vector_type(16))) __bf16 v16bf;
typedef __attribute__((ext_vector_type(8)))  __bf16 v8bf;
typedef __attribute__((ext_vector_type(8)))  float  v8f;

__device__ __forceinline__ unsigned short f2bf(float f) {
  union { float f; unsigned u; } x; x.f = f;
  unsigned r = x.u + 0x7FFFu + ((x.u >> 16) & 1u);   // round-to-nearest-even
  return (unsigned short)(r >> 16);
}

// ---------- prep kernels ----------------------------------------------------

__global__ void prep_dec(const float* __restrict__ dec_z,
                         const float* __restrict__ W_dec,
                         float* __restrict__ dp) {
  int idx = blockIdx.x * blockDim.x + threadIdx.x;          // B*ATT
  if (idx >= B_ * ATT_) return;
  int b = idx >> 9, a = idx & (ATT_ - 1);
  const float* z = dec_z + b * DUNITS_;
  float s = 0.f;
  for (int k = 0; k < DUNITS_; ++k) s += z[k] * W_dec[k * ATT_ + a];
  dp[idx] = s;
}

__global__ void prep_conv(const float* __restrict__ ap,
                          const float* __restrict__ cw,
                          float* __restrict__ out) {
  int idx = blockIdx.x * blockDim.x + threadIdx.x;          // B*CCH*T
  if (idx >= B_ * CCH_ * T_) return;
  int t = idx % T_;
  int c = (idx / T_) % CCH_;
  int b = idx / (T_ * CCH_);
  const float* apb = ap + (size_t)b * T_;
  const float* k = cw + c * KF_;
  float s = 0.f;
  int lo = t - PADF_;
  for (int kk = 0; kk < KF_; ++kk) {
    int tt = lo + kk;
    if (tt >= 0 && tt < T_) s += apb[tt] * k[kk];
  }
  out[idx] = s;
}

// Wt[n][k] = bf16(W_enc[k][n])  -> B-fragments become contiguous lane loads
__global__ void prep_wt(const float* __restrict__ W_enc,
                        unsigned short* __restrict__ Wt) {
  int idx = blockIdx.x * blockDim.x + threadIdx.x;          // ATT*EPROJS
  if (idx >= ATT_ * EPROJS_) return;
  int n = idx >> 10, k = idx & (EPROJS_ - 1);
  Wt[idx] = f2bf(W_enc[k * ATT_ + n]);
}

// ---------- main fused WMMA kernel: e[b,t] ---------------------------------
// grid (T/16, B), 256 threads = 8 waves. Wave w owns output cols [64w,64w+64).

__global__ __launch_bounds__(256) void attloc_main(
    const float* __restrict__ enc, const int* __restrict__ len,
    const float* __restrict__ b_enc, const float* __restrict__ W_att,
    const float* __restrict__ w_g, const float* __restrict__ b_g,
    const float* __restrict__ dec_proj, const float* __restrict__ convbuf,
    const unsigned short* __restrict__ Wt, float* __restrict__ e_out) {
  __shared__ unsigned short lds_enc[16 * LDSROW];
  __shared__ float conv_s[CCH_ * 16];
  __shared__ float ered[8 * 16];

  const int t0  = blockIdx.x * 16;
  const int b   = blockIdx.y;
  const int tid = threadIdx.x;

  // stage 16x1024 f32 tile -> bf16 LDS (tile rows are contiguous in memory)
  const float* encb = enc + ((size_t)b * T_ + t0) * EPROJS_;
  #pragma unroll
  for (int it = 0; it < 16; ++it) {
    int i = (tid + it * 256) * 4;
    float4 v = *(const float4*)(encb + i);
    int row = i >> 10, col = i & (EPROJS_ - 1);
    ushort4 u;
    u.x = f2bf(v.x); u.y = f2bf(v.y); u.z = f2bf(v.z); u.w = f2bf(v.w);
    *(ushort4*)(&lds_enc[row * LDSROW + col]) = u;
  }
  if (tid < CCH_ * 16) {
    int c = tid >> 4, m = tid & 15;
    conv_s[tid] = convbuf[((size_t)b * CCH_ + c) * T_ + t0 + m];
  }
  __syncthreads();

  const int wv   = tid >> 5;
  const int lane = tid & 31;
  const int hi   = lane >> 4;       // lane half selects K sub-ranges
  const int l15  = lane & 15;

  v8f acc[4] = {};
  // A frag (16x32 bf16): lane half hi reads contiguous K range [16hi,16hi+16)
  const unsigned short* lA = &lds_enc[l15 * LDSROW + 16 * hi];
  // B frag (32x16 bf16): lane (col l15) reads contiguous K range [16hi,16hi+16)
  const unsigned short* Wrow = Wt + (size_t)(wv * 64 + l15) * EPROJS_ + 16 * hi;

  for (int kb = 0; kb < EPROJS_; kb += 32) {
    v8bf alo = *(const v8bf*)(lA + kb);
    v8bf ahi = *(const v8bf*)(lA + kb + 8);
    v16bf a = __builtin_shufflevector(alo, ahi,
                0,1,2,3,4,5,6,7,8,9,10,11,12,13,14,15);
    #pragma unroll
    for (int nt = 0; nt < 4; ++nt) {
      v16bf bf = *(const v16bf*)(Wrow + (size_t)nt * 16 * EPROJS_ + kb);
      acc[nt] = __builtin_amdgcn_wmma_f32_16x16x32_bf16(
          false, a, false, bf, (short)0, acc[nt], false, false);
    }
  }

  // epilogue: + b_enc + dec_proj + conv x W_att, tanh, dot with w_g
  float ep[8] = {0.f,0.f,0.f,0.f,0.f,0.f,0.f,0.f};
  #pragma unroll
  for (int nt = 0; nt < 4; ++nt) {
    int a_idx = wv * 64 + nt * 16 + l15;
    float adda = b_enc[a_idx] + dec_proj[b * ATT_ + a_idx];
    float wg = w_g[a_idx];
    float wa[CCH_];
    #pragma unroll
    for (int c = 0; c < CCH_; ++c) wa[c] = W_att[c * ATT_ + a_idx];
    #pragma unroll
    for (int i = 0; i < 8; ++i) {
      int M = i + 8 * hi;                 // C/D layout: VGPR i -> row i+8*hi
      float x = acc[nt][i] + adda;
      #pragma unroll
      for (int c = 0; c < CCH_; ++c) x += conv_s[c * 16 + M] * wa[c];
      ep[i] += tanhf(x) * wg;
    }
  }
  // reduce across the 16 lanes of each half (same rows, different cols)
  #pragma unroll
  for (int m = 1; m < 16; m <<= 1) {
    #pragma unroll
    for (int i = 0; i < 8; ++i) ep[i] += __shfl_xor(ep[i], m, 32);
  }
  if (l15 == 0) {
    #pragma unroll
    for (int i = 0; i < 8; ++i) ered[wv * 16 + i + 8 * hi] = ep[i];
  }
  __syncthreads();
  if (tid < 16) {
    float s = 0.f;
    #pragma unroll
    for (int w = 0; w < 8; ++w) s += ered[w * 16 + tid];
    s += b_g[0];
    int t = t0 + tid;
    if (t >= len[b]) s = -INFINITY;
    e_out[(size_t)b * T_ + t] = s;
  }
}

// ---------- softmax over T (SCALING=2) -------------------------------------

__global__ __launch_bounds__(256) void attloc_softmax(
    const float* __restrict__ e, float* __restrict__ w) {
  __shared__ float red[256];
  int b = blockIdx.x, tid = threadIdx.x;
  const float* eb = e + (size_t)b * T_;
  float* wb = w + (size_t)b * T_;
  float mx = -INFINITY;
  for (int t = tid; t < T_; t += 256) mx = fmaxf(mx, eb[t]);
  red[tid] = mx; __syncthreads();
  for (int s = 128; s > 0; s >>= 1) {
    if (tid < s) red[tid] = fmaxf(red[tid], red[tid + s]);
    __syncthreads();
  }
  mx = red[0]; __syncthreads();
  float sum = 0.f;
  for (int t = tid; t < T_; t += 256) {
    float v = expf(2.0f * (eb[t] - mx));
    wb[t] = v; sum += v;
  }
  red[tid] = sum; __syncthreads();
  for (int s = 128; s > 0; s >>= 1) {
    if (tid < s) red[tid] += red[tid + s];
    __syncthreads();
  }
  float inv = 1.0f / red[0];
  for (int t = tid; t < T_; t += 256) wb[t] *= inv;
}

// ---------- context c = sum_t w[b,t]*enc[b,t,:] (deterministic 2-stage) ----

__global__ __launch_bounds__(256) void attloc_ctx_partial(
    const float* __restrict__ enc, const float* __restrict__ w,
    float* __restrict__ part) {
  int ex = blockIdx.x;            // 0..3 column chunk
  int tc = blockIdx.y;            // 0..7 time chunk (250 each)
  int b  = blockIdx.z;
  int col = ex * 256 + threadIdx.x;
  const float* encb = enc + (size_t)b * T_ * EPROJS_;
  const float* wb   = w + (size_t)b * T_;
  float s = 0.f;
  int t0 = tc * 250;
  for (int t = t0; t < t0 + 250; ++t)
    s += encb[(size_t)t * EPROJS_ + col] * wb[t];
  part[((size_t)b * 8 + tc) * EPROJS_ + col] = s;
}

__global__ void attloc_ctx_final(const float* __restrict__ part,
                                 float* __restrict__ c) {
  int idx = blockIdx.x * blockDim.x + threadIdx.x;   // B*EPROJS
  if (idx >= B_ * EPROJS_) return;
  int b = idx >> 10, col = idx & (EPROJS_ - 1);
  float s = 0.f;
  #pragma unroll
  for (int tc = 0; tc < 8; ++tc)
    s += part[((size_t)b * 8 + tc) * EPROJS_ + col];
  c[idx] = s;
}

// ---------------------------------------------------------------------------

extern "C" void kernel_launch(void* const* d_in, const int* in_sizes, int n_in,
                              void* d_out, int out_size, void* d_ws, size_t ws_size,
                              hipStream_t stream) {
  const float* enc   = (const float*)d_in[0];
  const int*   len   = (const int*)d_in[1];
  const float* dec_z = (const float*)d_in[2];
  const float* ap    = (const float*)d_in[3];
  const float* W_enc = (const float*)d_in[4];
  const float* b_enc = (const float*)d_in[5];
  const float* W_dec = (const float*)d_in[6];
  const float* W_att = (const float*)d_in[7];
  const float* cw    = (const float*)d_in[8];
  const float* w_g   = (const float*)d_in[9];
  const float* b_g   = (const float*)d_in[10];

  float* c_out = (float*)d_out;                  // [B, EPROJS]
  float* w_out = c_out + B_ * EPROJS_;           // [B, T]

  // workspace layout (floats, then bf16 Wt; ~5 MB total)
  float* ws    = (float*)d_ws;
  float* dp    = ws;                             // B*ATT        = 16384
  float* conv  = dp + B_ * ATT_;                 // B*CCH*T      = 640000
  float* e_buf = conv + B_ * CCH_ * T_;          // B*T          = 64000
  float* part  = e_buf + B_ * T_;                // B*8*EPROJS   = 262144
  unsigned short* Wt = (unsigned short*)(part + B_ * 8 * EPROJS_); // 1 MB, 32B-aligned

  prep_dec <<<(B_ * ATT_ + 255) / 256, 256, 0, stream>>>(dec_z, W_dec, dp);
  prep_conv<<<(B_ * CCH_ * T_ + 255) / 256, 256, 0, stream>>>(ap, cw, conv);
  prep_wt  <<<(ATT_ * EPROJS_ + 255) / 256, 256, 0, stream>>>(W_enc, Wt);

  attloc_main<<<dim3(T_ / 16, B_), 256, 0, stream>>>(
      enc, len, b_enc, W_att, w_g, b_g, dp, conv, Wt, e_buf);

  attloc_softmax<<<B_, 256, 0, stream>>>(e_buf, w_out);
  attloc_ctx_partial<<<dim3(4, 8, B_), 256, 0, stream>>>(enc, w_out, part);
  attloc_ctx_final<<<(B_ * EPROJS_ + 255) / 256, 256, 0, stream>>>(part, c_out);
}
